// MultiHeadAttention_26714696581555
// MI455X (gfx1250) — compile-verified, matrix operand order checked by reading, not run on hardware
//
#include <hip/hip_runtime.h>
#include <hip/hip_bf16.h>

// MHA for MI455X (gfx1250): bf16 WMMA everywhere, flash-style attention,
// software-pipelined (ping-pong) GEMM loops so WMMAs never wait on
// same-iteration loads.
// D_MODEL=1024, N_HEADS=16, D_HEAD=64, B=2, S=2048, TOKENS=4096.

typedef __attribute__((ext_vector_type(16))) __bf16 bf16x16;
typedef __attribute__((ext_vector_type(8)))  float  f32x8;
typedef __attribute__((ext_vector_type(4)))  int    i32x4;
typedef __attribute__((ext_vector_type(4)))  float  f32x4;

union ABu { bf16x16 v; i32x4 q[2]; int i[8]; };
struct Raw4 { f32x4 a[4]; };

__device__ __forceinline__ f32x8 wmma_bf16(bf16x16 a, bf16x16 b, f32x8 c) {
  // D = A(16x32 bf16) x B(32x16 bf16) + C(16x16 f32)
  return __builtin_amdgcn_wmma_f32_16x16x32_bf16(false, a, false, b, (short)0, c,
                                                 false, false);
}

__device__ __forceinline__ unsigned rne_bits(float f) {
  unsigned u = __float_as_uint(f);
  return u + 0x7FFFu + ((u >> 16) & 1u);   // round-to-nearest-even to bf16
}
__device__ __forceinline__ unsigned short bf16_rne(float f) {
  return (unsigned short)(rne_bits(f) >> 16);
}
// pack two fp32 -> packed bf16x2 dword with one v_perm_b32
__device__ __forceinline__ int pk_bf16(float lo, float hi) {
  return (int)__builtin_amdgcn_perm(rne_bits(hi), rne_bits(lo), 0x07060302u);
}
__device__ __forceinline__ void lds_fence() {
  asm volatile("s_wait_dscnt 0" ::: "memory");  // DS in-order per wave; fence compiler too
}

// ---------------------------------------------------------------------------
// Kernel 0: convert + transpose weights fp32[k][n] -> bf16 Wt[mat][n][k]
// ---------------------------------------------------------------------------
__global__ void k_wt(const float* __restrict__ Wq, const float* __restrict__ Wk,
                     const float* __restrict__ Wv, const float* __restrict__ Wo,
                     unsigned short* __restrict__ Wt) {
  int t = blockIdx.x * blockDim.x + threadIdx.x;   // 0 .. 4*1024*1024-1
  int mat = t >> 20;
  int r   = t & 0xFFFFF;
  int n   = r & 1023;    // output feature (coalesced read of row k)
  int k   = r >> 10;     // input feature
  const float* W = (mat == 0) ? Wq : (mat == 1) ? Wk : (mat == 2) ? Wv : Wo;
  Wt[(size_t)mat * 1048576 + (size_t)n * 1024 + k] = bf16_rne(W[(size_t)k * 1024 + n]);
}

// ---------------------------------------------------------------------------
// Kernel 1: fused QKV projection.  One wave = 16x64 output strip.
// p=0 -> Q [B,H,S,Dh]; p=1 -> K [B,H,S,Dh]; p=2 -> V^T [B,H,Dh,S]  (all bf16)
// ---------------------------------------------------------------------------
__global__ void __launch_bounds__(128) k_qkv(
    const float* __restrict__ X, const unsigned short* __restrict__ Wt,
    const float* __restrict__ bq, const float* __restrict__ bk,
    const float* __restrict__ bv,
    unsigned short* __restrict__ Qb, unsigned short* __restrict__ Kb,
    unsigned short* __restrict__ Vt) {
  const int lane   = threadIdx.x & 31;
  const int w      = blockIdx.x * 4 + (threadIdx.x >> 5);
  const int p      = w >> 12;       // 0..2
  const int rem    = w & 4095;
  const int mtile  = rem >> 4;      // 0..255 (token tiles)
  const int nstrip = rem & 15;      // 0..15  (64-col strips)
  const int row    = lane & 15;
  const int hi8    = (lane >> 4) << 3;   // K-half select for 16-bit A/B layout

  const float* Arow = X + (size_t)(mtile * 16 + row) * 1024;
  const unsigned short* Bbase = Wt + (size_t)p * 1048576;
  const unsigned short* Bn[4];
#pragma unroll
  for (int nt = 0; nt < 4; ++nt)
    Bn[nt] = Bbase + (size_t)(nstrip * 64 + nt * 16 + row) * 1024;

  auto loadAraw = [&](Raw4& r, int k0) {
    const float* ap = Arow + k0 + hi8;
    r.a[0] = *(const f32x4*)(ap);
    r.a[1] = *(const f32x4*)(ap + 4);
    r.a[2] = *(const f32x4*)(ap + 16);
    r.a[3] = *(const f32x4*)(ap + 20);
  };
  auto packA = [&](ABu& A, const Raw4& r) {
    A.i[0] = pk_bf16(r.a[0].x, r.a[0].y);  A.i[1] = pk_bf16(r.a[0].z, r.a[0].w);
    A.i[2] = pk_bf16(r.a[1].x, r.a[1].y);  A.i[3] = pk_bf16(r.a[1].z, r.a[1].w);
    A.i[4] = pk_bf16(r.a[2].x, r.a[2].y);  A.i[5] = pk_bf16(r.a[2].z, r.a[2].w);
    A.i[6] = pk_bf16(r.a[3].x, r.a[3].y);  A.i[7] = pk_bf16(r.a[3].z, r.a[3].w);
  };
  auto loadB = [&](ABu* Bt, int k0) {
#pragma unroll
    for (int nt = 0; nt < 4; ++nt) {
      const unsigned short* bp = Bn[nt] + k0 + hi8;
      Bt[nt].q[0] = *(const i32x4*)(bp);
      Bt[nt].q[1] = *(const i32x4*)(bp + 16);
    }
  };

  f32x8 zv = {0.f,0.f,0.f,0.f,0.f,0.f,0.f,0.f};
  f32x8 acc[4] = {zv, zv, zv, zv};

  // ping-pong software pipeline: loads for one half-step fly under the
  // other half-step's WMMAs
  Raw4 r0, r1; ABu B0[4], B1[4], A;
  loadAraw(r0, 0); loadB(B0, 0);
  for (int k0 = 0; k0 < 1024; k0 += 64) {
    loadAraw(r1, k0 + 32); loadB(B1, k0 + 32);
    packA(A, r0);
#pragma unroll
    for (int nt = 0; nt < 4; ++nt) acc[nt] = wmma_bf16(A.v, B0[nt].v, acc[nt]);
    if (k0 + 64 < 1024) { loadAraw(r0, k0 + 64); loadB(B0, k0 + 64); }
    packA(A, r1);
#pragma unroll
    for (int nt = 0; nt < 4; ++nt) acc[nt] = wmma_bf16(A.v, B1[nt].v, acc[nt]);
  }

  const float* bias = (p == 0) ? bq : (p == 1) ? bk : bv;
  unsigned short* dst01 = (p == 0) ? Qb : Kb;
#pragma unroll
  for (int nt = 0; nt < 4; ++nt) {
    const int n  = nstrip * 64 + nt * 16 + row;
    const float bz = bias[n];
    const int h = n >> 6, dh = n & 63;
#pragma unroll
    for (int r = 0; r < 8; ++r) {
      const int m = mtile * 16 + r + hi8;     // C-layout: VGPR r <-> row r + 8*hi
      const int b = m >> 11, s = m & 2047;
      const float val = acc[nt][r] + bz;
      if (p < 2) {
        dst01[(((size_t)(b * 16 + h) * 2048 + s) << 6) + dh] = bf16_rne(val);
      } else {
        Vt[(((size_t)(b * 16 + h) * 64 + dh) << 11) + s] = bf16_rne(val);
      }
    }
  }
}

// ---------------------------------------------------------------------------
// Kernel 2: flash attention.  One wave = one 16-row q-tile of one (b,h).
// K tiles double-buffered across j; V tiles loaded before softmax so their
// latency hides under the exp2/shuffle VALU stretch.
// ---------------------------------------------------------------------------
__global__ void __launch_bounds__(128) k_attn(
    const unsigned short* __restrict__ Qb, const unsigned short* __restrict__ Kb,
    const unsigned short* __restrict__ Vt, unsigned short* __restrict__ Ob) {
  __shared__ alignas(16) unsigned short Plds[4][16][40];  // 16B-aligned rows, low-conflict
  const int lane = threadIdx.x & 31;
  const int widx = threadIdx.x >> 5;
  const int w    = blockIdx.x * 4 + widx;
  const int bh   = w >> 7;     // 0..31
  const int qt   = w & 127;    // 0..127
  const int row  = lane & 15;
  const int hi8  = (lane >> 4) << 3;

  // Q as two 16x32 A-tiles, loaded once
  const unsigned short* Qrow = Qb + (((size_t)bh * 2048 + qt * 16 + row) << 6);
  bf16x16 qa[2];
#pragma unroll
  for (int kt = 0; kt < 2; ++kt) {
    ABu u;
    u.q[0] = *(const i32x4*)(Qrow + kt * 32 + hi8);
    u.q[1] = *(const i32x4*)(Qrow + kt * 32 + hi8 + 16);
    qa[kt] = u.v;
  }
  const unsigned short* Kh = Kb + ((size_t)bh << 17);   // [S][64]
  const unsigned short* Vh = Vt + ((size_t)bh << 17);   // [64][S]

  f32x8 zv = {0.f,0.f,0.f,0.f,0.f,0.f,0.f,0.f};
  f32x8 o[4] = {zv, zv, zv, zv};
  float mi[8], li[8];
#pragma unroll
  for (int r = 0; r < 8; ++r) { mi[r] = -1e30f; li[r] = 0.f; }

  const float SC = 0.125f * 1.44269504088896340736f;  // 1/sqrt(Dh) * log2(e)

  // kb[kt*2 + tile]: B-tiles of K for score columns j+tile*16 .. +15
  auto loadK = [&](ABu* kb, int j) {
#pragma unroll
    for (int kt = 0; kt < 2; ++kt)
#pragma unroll
      for (int t = 0; t < 2; ++t) {
        const unsigned short* kp =
            Kh + (((size_t)(j + t * 16 + row)) << 6) + kt * 32 + hi8;
        kb[kt * 2 + t].q[0] = *(const i32x4*)(kp);
        kb[kt * 2 + t].q[1] = *(const i32x4*)(kp + 16);
      }
  };

  auto step = [&](int j, ABu* kb) {
    // issue V-tile loads early: consumed only after softmax
    ABu vt[4];
#pragma unroll
    for (int nt = 0; nt < 4; ++nt) {
      const unsigned short* vp = Vh + (((size_t)(nt * 16 + row)) << 11) + j + hi8;
      vt[nt].q[0] = *(const i32x4*)(vp);
      vt[nt].q[1] = *(const i32x4*)(vp + 16);
    }
    // scores: columns j..j+15 (s0) and j+16..j+31 (s1)
    f32x8 s0 = zv, s1 = zv;
    s0 = wmma_bf16(qa[0], kb[0].v, s0);
    s1 = wmma_bf16(qa[0], kb[1].v, s1);
    s0 = wmma_bf16(qa[1], kb[2].v, s0);
    s1 = wmma_bf16(qa[1], kb[3].v, s1);
    s0 *= SC;  s1 *= SC;   // work in log2 domain

    // online softmax: each 16-lane half holds rows r+8*hi entirely
#pragma unroll
    for (int r = 0; r < 8; ++r) {
      float t = fmaxf(s0[r], s1[r]);
      t = fmaxf(t, __shfl_xor(t, 1, 32));
      t = fmaxf(t, __shfl_xor(t, 2, 32));
      t = fmaxf(t, __shfl_xor(t, 4, 32));
      t = fmaxf(t, __shfl_xor(t, 8, 32));
      const float mnew  = fmaxf(mi[r], t);
      const float alpha = exp2f(mi[r] - mnew);
      const float p0 = exp2f(s0[r] - mnew);
      const float p1 = exp2f(s1[r] - mnew);
      Plds[widx][r + hi8][row]      = bf16_rne(p0);
      Plds[widx][r + hi8][row + 16] = bf16_rne(p1);
      float rs = p0 + p1;
      rs += __shfl_xor(rs, 1, 32);
      rs += __shfl_xor(rs, 2, 32);
      rs += __shfl_xor(rs, 4, 32);
      rs += __shfl_xor(rs, 8, 32);
      li[r] = li[r] * alpha + rs;
      mi[r] = mnew;
      o[0][r] *= alpha; o[1][r] *= alpha; o[2][r] *= alpha; o[3][r] *= alpha;
    }
    lds_fence();
    // reload P as a 16x32 A-tile (C->A relayout via LDS)
    const unsigned short* prow = &Plds[widx][row][0];
    ABu pa;
    pa.q[0] = *(const i32x4*)(prow + hi8);
    pa.q[1] = *(const i32x4*)(prow + hi8 + 16);
    // O += P(16x32) x V(32x64); V tiles already in registers
#pragma unroll
    for (int nt = 0; nt < 4; ++nt) o[nt] = wmma_bf16(pa.v, vt[nt].v, o[nt]);
    lds_fence();
  };

  ABu kb0[4], kb1[4];
  loadK(kb0, 0);
  for (int j = 0; j < 2048; j += 64) {
    loadK(kb1, j + 32);                    // in flight under step(j)
    step(j, kb0);
    if (j + 64 < 2048) loadK(kb0, j + 64); // in flight under step(j+32)
    step(j + 32, kb1);
  }

  // normalize and write O as bf16 [token][H*Dh]
  const int b = bh >> 4, h = bh & 15;
#pragma unroll
  for (int r = 0; r < 8; ++r) {
    const float inv = 1.0f / li[r];
    const int tok = b * 2048 + qt * 16 + r + hi8;
#pragma unroll
    for (int nt = 0; nt < 4; ++nt) {
      const int col = h * 64 + nt * 16 + row;
      Ob[(size_t)tok * 1024 + col] = bf16_rne(o[nt][r] * inv);
    }
  }
}

// ---------------------------------------------------------------------------
// Kernel 3: output projection  out = O @ Wo + bo   (fp32 output)
// ---------------------------------------------------------------------------
__global__ void __launch_bounds__(128) k_oproj(
    const unsigned short* __restrict__ Ob, const unsigned short* __restrict__ Wt,
    const float* __restrict__ bo, float* __restrict__ out) {
  const int lane   = threadIdx.x & 31;
  const int w      = blockIdx.x * 4 + (threadIdx.x >> 5);
  const int mtile  = w >> 4;
  const int nstrip = w & 15;
  const int row    = lane & 15;
  const int hi8    = (lane >> 4) << 3;

  const unsigned short* Arow = Ob + (size_t)(mtile * 16 + row) * 1024;
  const unsigned short* WoT  = Wt + (size_t)3 * 1048576;
  const unsigned short* Bn[4];
#pragma unroll
  for (int nt = 0; nt < 4; ++nt)
    Bn[nt] = WoT + (size_t)(nstrip * 64 + nt * 16 + row) * 1024;

  auto loadA = [&](ABu& A, int k0) {
    const unsigned short* ap = Arow + k0 + hi8;
    A.q[0] = *(const i32x4*)(ap);
    A.q[1] = *(const i32x4*)(ap + 16);
  };
  auto loadB = [&](ABu* Bt, int k0) {
#pragma unroll
    for (int nt = 0; nt < 4; ++nt) {
      const unsigned short* bp = Bn[nt] + k0 + hi8;
      Bt[nt].q[0] = *(const i32x4*)(bp);
      Bt[nt].q[1] = *(const i32x4*)(bp + 16);
    }
  };

  f32x8 zv = {0.f,0.f,0.f,0.f,0.f,0.f,0.f,0.f};
  f32x8 acc[4] = {zv, zv, zv, zv};

  ABu A0, A1, B0[4], B1[4];
  loadA(A0, 0); loadB(B0, 0);
  for (int k0 = 0; k0 < 1024; k0 += 64) {
    loadA(A1, k0 + 32); loadB(B1, k0 + 32);
#pragma unroll
    for (int nt = 0; nt < 4; ++nt) acc[nt] = wmma_bf16(A0.v, B0[nt].v, acc[nt]);
    if (k0 + 64 < 1024) { loadA(A0, k0 + 64); loadB(B0, k0 + 64); }
#pragma unroll
    for (int nt = 0; nt < 4; ++nt) acc[nt] = wmma_bf16(A1.v, B1[nt].v, acc[nt]);
  }
#pragma unroll
  for (int nt = 0; nt < 4; ++nt) {
    const int n = nstrip * 64 + nt * 16 + row;
    const float bz = bo[n];
#pragma unroll
    for (int r = 0; r < 8; ++r) {
      out[(size_t)(mtile * 16 + r + hi8) * 1024 + n] = acc[nt][r] + bz;
    }
  }
}

// ---------------------------------------------------------------------------
extern "C" void kernel_launch(void* const* d_in, const int* in_sizes, int n_in,
                              void* d_out, int out_size, void* d_ws, size_t ws_size,
                              hipStream_t stream) {
  const float* x  = (const float*)d_in[0];
  const float* Wq = (const float*)d_in[1];
  const float* bq = (const float*)d_in[2];
  const float* Wk = (const float*)d_in[3];
  const float* bk = (const float*)d_in[4];
  const float* Wv = (const float*)d_in[5];
  const float* bv = (const float*)d_in[6];
  const float* Wo = (const float*)d_in[7];
  const float* bo = (const float*)d_in[8];
  float* out = (float*)d_out;

  char* ws = (char*)d_ws;
  unsigned short* Wt = (unsigned short*)(ws);                       //  8 MB: bf16 W^T x4
  unsigned short* Qb = (unsigned short*)(ws + ((size_t)8  << 20));  //  8 MB: Q  [B,H,S,Dh]
  unsigned short* Kb = (unsigned short*)(ws + ((size_t)16 << 20));  //  8 MB: K  [B,H,S,Dh]
  unsigned short* Vt = (unsigned short*)(ws + ((size_t)24 << 20));  //  8 MB: V^T[B,H,Dh,S]
  unsigned short* Ob = (unsigned short*)(ws + ((size_t)32 << 20));  //  8 MB: O  [tok,1024]

  k_wt   <<<16384, 256, 0, stream>>>(Wq, Wk, Wv, Wo, Wt);
  k_qkv  <<<3072,  128, 0, stream>>>(x, Wt, bq, bk, bv, Qb, Kb, Vt);
  k_attn <<<1024,  128, 0, stream>>>(Qb, Kb, Vt, Ob);
  k_oproj<<<1024,  128, 0, stream>>>(Ob, Wt, bo, out);
}
